// LearnedSparseScalarObservationFromNeighbors_36550171689197
// MI455X (gfx1250) — matrix operand order repaired and checked by reading, not compiled, not run on hardware
//
#include <hip/hip_runtime.h>
#include <hip/hip_bf16.h>
#include <math.h>

typedef __attribute__((ext_vector_type(16))) _Float16 v16h;
typedef __attribute__((ext_vector_type(8)))  _Float16 v8h;
typedef __attribute__((ext_vector_type(8)))  float    v8f;

#define NLON 1440
#define NLAT 721
#define CF   64        // feature channels (WMMA K)
#define HD   128       // hidden dim
#define QB   128       // queries per block
#define NS   3         // output scalars

// Branch-free tanh: prefer gfx1250 native TANH transcendental, else native-exp identity.
__device__ __forceinline__ float fast_tanhf(float y) {
#if __has_builtin(__builtin_amdgcn_tanhf)
    return __builtin_amdgcn_tanhf(y);
#elif __has_builtin(__builtin_amdgcn_tanh_f32)
    return __builtin_amdgcn_tanh_f32(y);
#else
    // tanh(|y|) = (1 - e) / (1 + e), e = exp(-2|y|)  -- straight-line, no EXEC branches
    float e = __expf(-2.0f * fabsf(y));
#if __has_builtin(__builtin_amdgcn_rcpf)
    float t = (1.0f - e) * __builtin_amdgcn_rcpf(1.0f + e);
#else
    float t = (1.0f - e) / (1.0f + e);
#endif
    return copysignf(t, y);
#endif
}

__global__ __launch_bounds__(256)
void fused_obs_kernel(const float* __restrict__ lon_grid,
                      const float* __restrict__ lat_grid,
                      const float* __restrict__ features,
                      const float* __restrict__ lon_q,
                      const float* __restrict__ lat_q,
                      const float* __restrict__ W1,
                      const float* __restrict__ b1,
                      const float* __restrict__ W2,
                      const float* __restrict__ b2,
                      float* __restrict__ out,
                      int nq)
{
    __shared__ __align__(32) _Float16 xs[QB * CF];    // gathered features, [q][k] f16
    __shared__ __align__(32) _Float16 w1t[HD * CF];   // W1 transposed (first 64 rows), [n][k] f16
    __shared__ float dlonS[QB], dlatS[QB];
    __shared__ int   liS[QB], laS[QB];
    __shared__ float w1r64[HD], w1r65[HD], b1S[HD];
    __shared__ float w2S[HD * NS];
    __shared__ float b2S[NS];

    const int tid    = threadIdx.x;
    const int lane   = tid & 31;
    const int laneLo = lane & 15;
    const bool hiH   = (lane >= 16);
    const int wave   = tid >> 5;               // 0..7 -> M-tile of 16 queries
    const int q0     = blockIdx.x * QB;

    // ---------------- Phase 1a: closed-form nearest-neighbor indices ----------------
    if (tid < QB) {
        int q = q0 + tid;
        if (q < nq) {
            float lon = lon_q[q];
            float lat = lat_q[q];
            int li = (int)rintf(lon * 4.0f);             // 0.25 deg uniform grid
            if (li >= NLON) li -= NLON;
            if (li < 0)     li += NLON;
            int la = (int)rintf((lat + 90.0f) * 4.0f);
            la = min(max(la, 0), NLAT - 1);
            liS[tid]   = li;
            laS[tid]   = la;
            dlonS[tid] = lon - lon_grid[li];
            dlatS[tid] = lat - lat_grid[la];
        } else {
            liS[tid] = 0; laS[tid] = 0; dlonS[tid] = 0.0f; dlatS[tid] = 0.0f;
        }
    }
    // stage small weight pieces (f32)
    for (int i = tid; i < HD; i += 256) {
        w1r64[i] = W1[64 * HD + i];
        w1r65[i] = W1[65 * HD + i];
        b1S[i]   = b1[i];
    }
    for (int i = tid; i < HD * NS; i += 256) w2S[i] = W2[i];
    if (tid < NS) b2S[tid] = b2[tid];
    // stage W1^T first 64 rows as f16: coalesced reads along n
    for (int i = tid; i < HD * CF; i += 256) {
        int k = i >> 7;            // 0..63
        int n = i & 127;           // 0..127
        w1t[n * CF + k] = (_Float16)W1[k * HD + n];
    }
    __syncthreads();

    // ---------------- Phase 1b: random gather of neighbor features ----------------
    for (int i = tid; i < QB * CF; i += 256) {
        int q = i >> 6;            // 0..127
        int c = i & 63;            // 0..63
        size_t off = (size_t)c * ((size_t)NLON * NLAT)
                   + (size_t)liS[q] * NLAT + (size_t)laS[q];
        xs[q * CF + c] = (_Float16)features[off];
    }
    __syncthreads();

    // ---------------- Phase 2: WMMA GEMM1 + GELU + in-register GEMM2 ----------------
    const int mrow = wave * 16;                 // local first query row of this wave
    const int kbase = hiH ? 8 : 0;              // ISA 16-bit A 16x32 layout

    // A fragments (K chunks 0..31 and 32..63), M = laneLo
    const _Float16* arow = &xs[(mrow + laneLo) * CF];
    v8h a0lo = *(const v8h*)(arow + kbase);
    v8h a0hi = *(const v8h*)(arow + kbase + 16);
    v16h a0  = __builtin_shufflevector(a0lo, a0hi, 0,1,2,3,4,5,6,7,8,9,10,11,12,13,14,15);
    v8h a1lo = *(const v8h*)(arow + 32 + kbase);
    v8h a1hi = *(const v8h*)(arow + 32 + kbase + 16);
    v16h a1  = __builtin_shufflevector(a1lo, a1hi, 0,1,2,3,4,5,6,7,8,9,10,11,12,13,14,15);

    float pout[8][NS];
    #pragma unroll
    for (int r = 0; r < 8; ++r)
        #pragma unroll
        for (int s = 0; s < NS; ++s) pout[r][s] = 0.0f;

    const int kboff = hiH ? 8 : 0;              // B layout: same table order as A (K 8h..8h+7, 16+8h..16+8h+7)

    #pragma unroll
    for (int j = 0; j < 8; ++j) {               // 8 N-tiles of 16 -> H = 128
        const int n = j * 16 + laneLo;
        v16h bf0 = __builtin_shufflevector(*(const v8h*)&w1t[n * CF + kboff], *(const v8h*)&w1t[n * CF + kboff + 16], 0,1,2,3,4,5,6,7,8,9,10,11,12,13,14,15);        // K chunk 0
        v16h bf1 = __builtin_shufflevector(*(const v8h*)&w1t[n * CF + 32 + kboff], *(const v8h*)&w1t[n * CF + 32 + kboff + 16], 0,1,2,3,4,5,6,7,8,9,10,11,12,13,14,15);   // K chunk 1

        // C init = b1 + dlon*W1[64] + dlat*W1[65]  (folds the 2 extra K columns)
        v8f c;
        const float bcol  = b1S[n];
        const float w64   = w1r64[n];
        const float w65   = w1r65[n];
        #pragma unroll
        for (int r = 0; r < 8; ++r) {
            int qrow = mrow + r + (hiH ? 8 : 0);
            c[r] = fmaf(dlonS[qrow], w64, fmaf(dlatS[qrow], w65, bcol));
        }

        c = __builtin_amdgcn_wmma_f32_16x16x32_f16(false, a0, false, bf0, (short)0, c, false, false);
        c = __builtin_amdgcn_wmma_f32_16x16x32_f16(false, a1, false, bf1, (short)0, c, false, false);

        // tanh-approx GELU (jax.nn.gelu default) + accumulate second GEMM partials
        #pragma unroll
        for (int r = 0; r < 8; ++r) {
            float x  = c[r];
            float t  = fast_tanhf(0.7978845608028654f * fmaf(0.044715f * x, x * x, x));
            float hg = 0.5f * x * (1.0f + t);
            #pragma unroll
            for (int s = 0; s < NS; ++s)
                pout[r][s] = fmaf(hg, w2S[n * NS + s], pout[r][s]);
        }
    }

    // ---------------- Phase 3: cross-lane reduce (16 lanes per half) + store ----------------
    #pragma unroll
    for (int r = 0; r < 8; ++r) {
        #pragma unroll
        for (int s = 0; s < NS; ++s) {
            float v = pout[r][s];
            v += __shfl_xor(v, 1, 32);
            v += __shfl_xor(v, 2, 32);
            v += __shfl_xor(v, 4, 32);
            v += __shfl_xor(v, 8, 32);
            if (laneLo == 0) {
                int q = q0 + mrow + r + (hiH ? 8 : 0);
                if (q < nq) out[q * NS + s] = v + b2S[s];
            }
        }
    }
}

extern "C" void kernel_launch(void* const* d_in, const int* in_sizes, int n_in,
                              void* d_out, int out_size, void* d_ws, size_t ws_size,
                              hipStream_t stream) {
    const float* lon_grid = (const float*)d_in[0];
    const float* lat_grid = (const float*)d_in[1];
    const float* features = (const float*)d_in[2];
    const float* lon_q    = (const float*)d_in[3];
    const float* lat_q    = (const float*)d_in[4];
    const float* W1       = (const float*)d_in[5];
    const float* b1       = (const float*)d_in[6];
    const float* W2       = (const float*)d_in[7];
    const float* b2       = (const float*)d_in[8];
    float* out            = (float*)d_out;

    const int nq = in_sizes[3];
    const int blocks = (nq + QB - 1) / QB;
    fused_obs_kernel<<<dim3(blocks), dim3(256), 0, stream>>>(
        lon_grid, lat_grid, features, lon_q, lat_q, W1, b1, W2, b2, out, nq);
}
